// MultiHeadAttention_65352222376626
// MI455X (gfx1250) — matrix operand order repaired and checked by reading, not compiled, not run on hardware
//
#include <hip/hip_runtime.h>
#include <hip/hip_bf16.h>

// ---------------------------------------------------------------------------
// Types for CDNA5 WMMA (wave32): v_wmma_f32_16x16x32_bf16
// ---------------------------------------------------------------------------
typedef __bf16 bf16;
typedef __attribute__((ext_vector_type(16))) __bf16 bf16x16;
typedef __attribute__((ext_vector_type(8)))  __bf16 bf16x8;
typedef __attribute__((ext_vector_type(4)))  __bf16 bf16x4;
typedef __attribute__((ext_vector_type(8)))  float  f32x8;
typedef __attribute__((ext_vector_type(4)))  int    v4i;

#define WMMA_BF16(a, b, c) \
  __builtin_amdgcn_wmma_f32_16x16x32_bf16(false, (a), false, (b), (short)0, (c), false, false)

#define AS1 __attribute__((address_space(1)))
#define AS3 __attribute__((address_space(3)))

// CDNA5 async global->LDS path (ASYNCcnt), guarded so absence cannot break
// the build; falls back to uint4 register staging.
#if defined(__AMDGCN__) && __has_builtin(__builtin_amdgcn_global_load_async_to_lds_b128) && \
    __has_builtin(__builtin_amdgcn_s_wait_asynccnt)
#define HAVE_ASYNC_LDS 1
#else
#define HAVE_ASYNC_LDS 0
#endif

__device__ __forceinline__ void copy_b128_to_lds(const void* gsrc, void* lds) {
#if HAVE_ASYNC_LDS
  __builtin_amdgcn_global_load_async_to_lds_b128((AS1 v4i*)gsrc, (AS3 v4i*)lds, 0, 0);
#else
  *(uint4*)lds = *(const uint4*)gsrc;
#endif
}
__device__ __forceinline__ void async_lds_wait() {
#if HAVE_ASYNC_LDS
  __builtin_amdgcn_s_wait_asynccnt(0);
#endif
}

constexpr int B_ = 8, S_ = 1024, D_ = 768, H_ = 12;
constexpr float SCORE_SCALE = 8.0f;  // sqrt(head_dim=64); reference multiplies by this

// ---------------------------------------------------------------------------
// Fragment helper. ISA 16-bit A layout (16x32, MxK): lanes 0-15 hold K = [0..7]
// then [16..23]; lanes 16-31 hold K = [8..15] then [24..31].  B layout (32x16,
// KxN stored as [N][K]): same K interleave as A.
// Both reduce to two contiguous 16B reads per lane.
// ---------------------------------------------------------------------------
__device__ __forceinline__ bf16x16 mk_frag(const bf16* p0, const bf16* p1) {
  bf16x8 lo = *(const bf16x8*)p0;
  bf16x8 hi = *(const bf16x8*)p1;
  bf16x16 r;
#pragma unroll
  for (int i = 0; i < 8; ++i) { r[i] = lo[i]; r[i + 8] = hi[i]; }
  return r;
}

// ---------------------------------------------------------------------------
// fp32 -> bf16 conversion (vectorized x4; all sizes are multiples of 1024)
// ---------------------------------------------------------------------------
__global__ __launch_bounds__(256) void cvt_f32_bf16(const float* __restrict__ in,
                                                    bf16* __restrict__ out, int n) {
  int i = (blockIdx.x * 256 + threadIdx.x) * 4;
  if (i + 3 < n) {
    float4 v = *(const float4*)(in + i);
    bf16x4 o;
    o[0] = (bf16)v.x; o[1] = (bf16)v.y; o[2] = (bf16)v.z; o[3] = (bf16)v.w;
    *(bf16x4*)(out + i) = o;
  }
}

// ---------------------------------------------------------------------------
// 128x128 WMMA GEMM tile core.  C = A[M,K] * B[K,N] (+bias).  A row-major,
// B row-major in memory, staged transposed [N][K] in LDS so B-fragments are
// contiguous.  8 waves each own a 16-row stripe and the full 128-col width.
// A tile staged via async-to-LDS when available; next tiles prefetched.
// ---------------------------------------------------------------------------
template <bool OUT_F32>
__device__ __forceinline__ void gemm_tile_128x128(
    const bf16* __restrict__ A, int lda, const bf16* __restrict__ B, int ldb,
    const float* __restrict__ bias, bf16* __restrict__ Cb, float* __restrict__ Cf,
    int ldc, int m0, int n0, int K) {
  __shared__ bf16 As[128][32];      // [M][K]
  __shared__ bf16 Bs[128][40];      // [N][K] transposed, padded (stride 80B = 5*16B)

  const int tid = threadIdx.x, lane = tid & 31, wid = tid >> 5;
  const int l16 = lane & 15;
  const int ah = (lane & 16) ? 8 : 0;   // A-fragment K base (ISA lane-half interleave)
  const int bh = (lane & 16) ? 16 : 0;  // unused; B fragments also use ah

  f32x8 acc[8];
#pragma unroll
  for (int nt = 0; nt < 8; ++nt)
#pragma unroll
    for (int r = 0; r < 8; ++r) acc[nt][r] = 0.f;

  const int arow = tid >> 1, aseg = tid & 1;

  for (int kt = 0; kt < K; kt += 32) {
    // A tile: 128x32 bf16 = 8KB; one 16B chunk per thread (async when present).
    {
      const bf16* src = A + (size_t)(m0 + arow) * lda + kt + aseg * 8;
      copy_b128_to_lds(src, &As[arow][aseg * 8]);
      if (kt + 32 < K)
        __builtin_prefetch(src + 32, 0, 0);  // global_prefetch_b8 for next K tile
    }
    // B tile: 32x128, stored transposed.  512 8-element chunks, 2 per thread.
#pragma unroll
    for (int i = 0; i < 2; ++i) {
      int idx = tid + 256 * i;
      int kk = idx >> 4, nseg = idx & 15;
      const bf16* src = B + (size_t)(kt + kk) * ldb + n0 + nseg * 8;
      bf16x8 v = *(const bf16x8*)src;
      if (kt + 32 < K)
        __builtin_prefetch(src + (size_t)32 * ldb, 0, 0);
#pragma unroll
      for (int e = 0; e < 8; ++e) Bs[nseg * 8 + e][kk] = v[e];
    }
    async_lds_wait();
    __syncthreads();

    {
      const int row = wid * 16 + l16;
      bf16x16 a = mk_frag(&As[row][ah], &As[row][ah + 16]);
      // Batch all B-fragment loads, then fire the 8 WMMAs back-to-back so a
      // single s_wait_dscnt covers the whole burst.
      bf16x16 bfr[8];
#pragma unroll
      for (int nt = 0; nt < 8; ++nt) {
        const bf16* bp = &Bs[nt * 16 + l16][ah];
        bfr[nt] = mk_frag(bp, bp + 16);
      }
#pragma unroll
      for (int nt = 0; nt < 8; ++nt) acc[nt] = WMMA_BF16(a, bfr[nt], acc[nt]);
    }
    __syncthreads();
  }

  // Epilogue: C layout — lane 0-15: M = r, lane 16-31: M = r + 8; N = lane%16.
  const int mrow = m0 + wid * 16 + ((lane & 16) ? 8 : 0);
#pragma unroll
  for (int nt = 0; nt < 8; ++nt) {
    int n = n0 + nt * 16 + l16;
    float bv = bias ? bias[n] : 0.f;
#pragma unroll
    for (int r = 0; r < 8; ++r) {
      float v = acc[nt][r] + bv;
      size_t idx = (size_t)(mrow + r) * ldc + n;
      if constexpr (OUT_F32) Cf[idx] = v; else Cb[idx] = (bf16)v;
    }
  }
}

// ---------------------------------------------------------------------------
// QKV projections: for each (proj, h, b): x[b] (1024x768) @ W[proj][h] (768x768)
// + bias -> {q,k,v}[b,h] (bf16).
// ---------------------------------------------------------------------------
__global__ __launch_bounds__(256) void qkv_gemm(
    const bf16* __restrict__ xb,
    const bf16* __restrict__ wq, const bf16* __restrict__ wk, const bf16* __restrict__ wv,
    const float* __restrict__ bq, const float* __restrict__ bk, const float* __restrict__ bv,
    bf16* __restrict__ q, bf16* __restrict__ k, bf16* __restrict__ v) {
  int tile = blockIdx.x;                // 48 = 8 Mtiles * 6 Ntiles
  int mtile = tile & 7, ntile = tile >> 3;
  int ph = blockIdx.y;                  // 36 = 3 proj * 12 heads
  int proj = ph / H_, h = ph % H_;
  int b = blockIdx.z;

  const bf16* W; const float* bias; bf16* out;
  if (proj == 0)      { W = wq; bias = bq; out = q; }
  else if (proj == 1) { W = wk; bias = bk; out = k; }
  else                { W = wv; bias = bv; out = v; }

  const bf16* Am = xb + (size_t)b * S_ * D_;
  const bf16* Bm = W + (size_t)h * D_ * D_;
  const float* bs = bias + h * D_;
  bf16* C = out + (size_t)(b * H_ + h) * S_ * D_;
  gemm_tile_128x128<false>(Am, D_, Bm, D_, bs, C, nullptr, D_,
                           mtile * 128, ntile * 128, D_);
}

// ---------------------------------------------------------------------------
// Output projection: o_concat [8192, 9216] @ Wp [9216, 768] + bp -> f32 out.
// ---------------------------------------------------------------------------
__global__ __launch_bounds__(256) void out_gemm(
    const bf16* __restrict__ ob, const bf16* __restrict__ wp,
    const float* __restrict__ bp, float* __restrict__ out) {
  gemm_tile_128x128<true>(ob, H_ * D_, wp, D_, bp, nullptr, out, D_,
                          blockIdx.x * 128, blockIdx.y * 128, H_ * D_);
}

// ---------------------------------------------------------------------------
// Flash attention (per-head width E=768, scores scaled by sqrt(64)).
// Block = 256 threads (8 waves) handles 32 query rows of one (b,h).
// Wave wid = mw*4 + ew:  stage1: (mw, nw=ew) tile of the 32x64 score block;
// stage2: (mw, ew) -> 16 rows x 192 output columns (12 WMMA accumulators).
// Q fragments live in registers; K in LDS [64][768] (async-to-LDS when
// available); V in LDS transposed [768][64].  Online softmax via per-row
// stats in static LDS.  Dynamic LDS: 192KB (within CDNA5's 320KB/WGP).
// ---------------------------------------------------------------------------
__global__ __launch_bounds__(256) void flash_attn(
    const bf16* __restrict__ qg, const bf16* __restrict__ kg,
    const bf16* __restrict__ vg, bf16* __restrict__ og) {
  constexpr int E = D_, NB = 64, MB = 32, KC = E / 32;  // KC = 24
  const int qb = blockIdx.x, h = blockIdx.y, b = blockIdx.z;
  const int tid = threadIdx.x, lane = tid & 31, wid = tid >> 5;
  const int l16 = lane & 15;
  const int mw = wid >> 2, ew = wid & 3, nw = wid & 3;
  const int ah = (lane & 16) ? 8 : 0;
  const int bh = (lane & 16) ? 16 : 0;

  extern __shared__ char dyn[];
  bf16* Ks = (bf16*)dyn;                                   // [NB][E]   96KB
  bf16* Vt = (bf16*)(dyn + (size_t)NB * E * sizeof(bf16)); // [E][NB]   96KB
  __shared__ float Sbuf[MB][NB + 1];
  __shared__ bf16  Pbuf[MB][NB + 8];
  __shared__ float mrun[MB], lrun[MB], scl[MB];

  const size_t bhoff = (size_t)(b * H_ + h) * S_ * E;
  const bf16* qbase = qg + bhoff;
  const bf16* kbase = kg + bhoff;
  const bf16* vbase = vg + bhoff;

  // Q fragments for this wave's 16 query rows (row = lane%16 within sub-tile).
  const bf16* qp = qbase + (size_t)(qb * MB + mw * 16 + l16) * E;
  bf16x16 qf[KC];
#pragma unroll
  for (int kk = 0; kk < KC; ++kk)
    qf[kk] = mk_frag(qp + kk * 32 + ah, qp + kk * 32 + ah + 16);

  f32x8 oacc[12];
#pragma unroll
  for (int t = 0; t < 12; ++t)
#pragma unroll
    for (int r = 0; r < 8; ++r) oacc[t][r] = 0.f;

  if (tid < MB) { mrun[tid] = -INFINITY; lrun[tid] = 0.f; }

  for (int kb = 0; kb < S_ / NB; ++kb) {
    __syncthreads();
    // K block: 64x768 bf16 = 6144 x 16B flat copy (async-to-LDS when present).
    {
      const bf16* src = kbase + (size_t)kb * NB * E;
      for (int i = tid; i < NB * E / 8; i += 256)
        copy_b128_to_lds(src + (size_t)i * 8, Ks + (size_t)i * 8);
    }
    // V block, transposed into [e][key] (register staging; transpose on store).
    {
      const bf16* src = vbase + (size_t)kb * NB * E;
      for (int i = tid; i < NB * (E / 8); i += 256) {
        int kk = i / 96, eseg = i - kk * 96;
        bf16x8 v = *(const bf16x8*)(src + (size_t)kk * E + eseg * 8);
#pragma unroll
        for (int e = 0; e < 8; ++e) Vt[(size_t)(eseg * 8 + e) * NB + kk] = v[e];
      }
    }
    async_lds_wait();
    __syncthreads();

    // Stage 1: 16x16 score tile per wave over K = 768, grouped 4 loads + 4 WMMAs.
    f32x8 sacc;
#pragma unroll
    for (int r = 0; r < 8; ++r) sacc[r] = 0.f;
#pragma unroll
    for (int g = 0; g < KC / 4; ++g) {
      bf16x16 kf[4];
#pragma unroll
      for (int j = 0; j < 4; ++j) {
        const bf16* kp = Ks + (size_t)(nw * 16 + l16) * E + (g * 4 + j) * 32;
        kf[j] = mk_frag(kp + ah, kp + ah + 16);
      }
#pragma unroll
      for (int j = 0; j < 4; ++j) sacc = WMMA_BF16(qf[g * 4 + j], kf[j], sacc);
    }
    {
      int row = mw * 16 + ((lane & 16) ? 8 : 0);
      int col = nw * 16 + l16;
#pragma unroll
      for (int r = 0; r < 8; ++r) Sbuf[row + r][col] = sacc[r] * SCORE_SCALE;
    }
    __syncthreads();

    // Online softmax: one thread per query row.
    if (tid < MB) {
      float mprev = mrun[tid], mx = mprev;
#pragma unroll 8
      for (int c = 0; c < NB; ++c) mx = fmaxf(mx, Sbuf[tid][c]);
      float rescale = __expf(mprev - mx);
      float sum = 0.f;
      for (int c = 0; c < NB; ++c) {
        float p = __expf(Sbuf[tid][c] - mx);
        sum += p;
        Pbuf[tid][c] = (bf16)p;
      }
      mrun[tid] = mx;
      lrun[tid] = lrun[tid] * rescale + sum;
      scl[tid] = rescale;
    }
    __syncthreads();

    // Rescale running O accumulators by exp(m_old - m_new) per row.
    {
      int rbase = mw * 16 + ((lane & 16) ? 8 : 0);
      float f[8];
#pragma unroll
      for (int r = 0; r < 8; ++r) f[r] = scl[rbase + r];
#pragma unroll
      for (int t = 0; t < 12; ++t)
#pragma unroll
        for (int r = 0; r < 8; ++r) oacc[t][r] *= f[r];
    }

    // Stage 2: O += P(16x64) @ V(64x192 slice); batch 4 B-frags per WMMA burst.
    {
      const bf16* prow = &Pbuf[mw * 16 + l16][0];
#pragma unroll
      for (int kk2 = 0; kk2 < 2; ++kk2) {
        bf16x16 a = mk_frag(prow + kk2 * 32 + ah, prow + kk2 * 32 + ah + 16);
#pragma unroll
        for (int g = 0; g < 3; ++g) {
          bf16x16 bfr[4];
#pragma unroll
          for (int j = 0; j < 4; ++j) {
            int nt = g * 4 + j;
            const bf16* vp = Vt + (size_t)(ew * 192 + nt * 16 + l16) * NB + kk2 * 32;
            bfr[j] = mk_frag(vp + ah, vp + ah + 16);
          }
#pragma unroll
          for (int j = 0; j < 4; ++j)
            oacc[g * 4 + j] = WMMA_BF16(a, bfr[j], oacc[g * 4 + j]);
        }
      }
    }
  }
  __syncthreads();

  // Epilogue: normalize by running sum, store head-concat layout [B,S,H*E].
  {
    int rbase = mw * 16 + ((lane & 16) ? 8 : 0);
    bf16* obase = og + (size_t)(b * S_ + qb * MB) * (H_ * E) + (size_t)h * E;
#pragma unroll
    for (int r = 0; r < 8; ++r) {
      float inv = 1.f / lrun[rbase + r];
      bf16* orow = obase + (size_t)(rbase + r) * (H_ * E);
#pragma unroll
      for (int nt = 0; nt < 12; ++nt) {
        int e = ew * 192 + nt * 16 + l16;
        orow[e] = (bf16)(oacc[nt][r] * inv);
      }
    }
  }
}

// ---------------------------------------------------------------------------
// Host launcher
// ---------------------------------------------------------------------------
extern "C" void kernel_launch(void* const* d_in, const int* in_sizes, int n_in,
                              void* d_out, int out_size, void* d_ws, size_t ws_size,
                              hipStream_t stream) {
  const float* x  = (const float*)d_in[0];
  const float* Wq = (const float*)d_in[1];
  const float* bq = (const float*)d_in[2];
  const float* Wk = (const float*)d_in[3];
  const float* bk = (const float*)d_in[4];
  const float* Wv = (const float*)d_in[5];
  const float* bv = (const float*)d_in[6];
  const float* Wp = (const float*)d_in[7];
  const float* bp = (const float*)d_in[8];
  float* out = (float*)d_out;

  const size_t nX  = (size_t)B_ * S_ * D_;        // 6,291,456
  const size_t nW  = (size_t)H_ * D_ * D_;        // 7,077,888
  const size_t nWp = (size_t)H_ * D_ * D_;        // 9216*768
  const size_t nQ  = (size_t)B_ * H_ * S_ * D_;   // 75,497,472

  size_t off = 0;
  auto alloc = [&](size_t elems) {
    bf16* p = (bf16*)((char*)d_ws + off);
    off += (elems * sizeof(bf16) + 255) & ~(size_t)255;
    return p;
  };
  bf16* xb  = alloc(nX);
  bf16* wqb = alloc(nW);
  bf16* wkb = alloc(nW);
  bf16* wvb = alloc(nW);
  bf16* wpb = alloc(nWp);
  bf16* qb  = alloc(nQ);
  bf16* kb  = alloc(nQ);
  bf16* vb  = alloc(nQ);
  bf16* ob  = alloc(nQ);
  (void)ws_size; (void)n_in; (void)in_sizes; (void)out_size;

  // 1) fp32 -> bf16 conversions
  cvt_f32_bf16<<<dim3((unsigned)(nX / 1024)),  256, 0, stream>>>(x,  xb,  (int)nX);
  cvt_f32_bf16<<<dim3((unsigned)(nW / 1024)),  256, 0, stream>>>(Wq, wqb, (int)nW);
  cvt_f32_bf16<<<dim3((unsigned)(nW / 1024)),  256, 0, stream>>>(Wk, wkb, (int)nW);
  cvt_f32_bf16<<<dim3((unsigned)(nW / 1024)),  256, 0, stream>>>(Wv, wvb, (int)nW);
  cvt_f32_bf16<<<dim3((unsigned)(nWp / 1024)), 256, 0, stream>>>(Wp, wpb, (int)nWp);

  // 2) QKV projections: grid (48 tiles, 3*12 proj/head, 8 batch)
  qkv_gemm<<<dim3(48, 36, B_), 256, 0, stream>>>(xb, wqb, wkb, wvb,
                                                 bq, bk, bv, qb, kb, vb);

  // 3) Flash attention: grid (32 query blocks, 12 heads, 8 batch), 192KB dyn LDS
  size_t flash_lds = (size_t)(64 * D_ + D_ * 64) * sizeof(bf16);
  flash_attn<<<dim3(S_ / 32, H_, B_), 256, flash_lds, stream>>>(qb, kb, vb, ob);

  // 4) Output projection: [8192,9216]@[9216,768] -> f32 out
  out_gemm<<<dim3((B_ * S_) / 128, D_ / 128, 1), 256, 0, stream>>>(ob, wpb, bp, out);
}
